// GPT2QuantAttention_52673478918863
// MI455X (gfx1250) — 16-bit and 8-bit matrix operand order repaired and checked by reading, compiled, not run on hardware
//
#include <hip/hip_runtime.h>
#include <hip/hip_bf16.h>
#include <stdint.h>

#define B_  2
#define S_  2048
#define D_  1024
#define H_  16
#define HD_ 64
#define M_  (B_ * S_)

typedef __attribute__((ext_vector_type(4)))  unsigned int v4u;
typedef __attribute__((ext_vector_type(4)))  int      v4i;
typedef __attribute__((ext_vector_type(8)))  int      v8i;
typedef __attribute__((ext_vector_type(8)))  float    v8f;
typedef __attribute__((ext_vector_type(16))) _Float16 v16h;

union U16H { v16h v; int i[8]; _Float16 h[16]; };
union U8I  { v8i  v; int i[8]; };

#if __has_builtin(__builtin_amdgcn_tensor_load_to_lds)
#define QATTN_HAS_TDM 1
#else
#define QATTN_HAS_TDM 0
#endif

#define GEMM_LDS_BYTES 12288
#define TDM_SCRATCH_OFF 4096u

__device__ __forceinline__ void qattn_tdm_warm(const void* g, int K, unsigned rows) {
#if QATTN_HAS_TDM
  const unsigned long long ga = (unsigned long long)(uintptr_t)g;
  const unsigned td0 = (unsigned)K;
  const unsigned td1 = rows;
  const unsigned tile0 = 64u;
  const unsigned tile1 = rows;
  const unsigned stride = (unsigned)K;

  v4u g0 = { 1u, TDM_SCRATCH_OFF, (unsigned)(ga & 0xFFFFFFFFu),
             (unsigned)((ga >> 32) & 0x01FFFFFFu) | (2u << 30) };
  v8i g1 = { 0,
             (int)((td0 & 0xFFFFu) << 16),
             (int)((td0 >> 16) | ((td1 & 0xFFFFu) << 16)),
             (int)((td1 >> 16) | (tile0 << 16)),
             (int)tile1,
             (int)stride,
             0, 0 };
  v4i z = { 0, 0, 0, 0 };
#if defined(__clang_major__) && (__clang_major__ >= 23)
  v8i z8 = {};
  __builtin_amdgcn_tensor_load_to_lds(g0, g1, z, z, z8, 0);
#else
  __builtin_amdgcn_tensor_load_to_lds(g0, g1, z, z, 0);
#endif
#endif
}

__device__ __forceinline__ void qattn_tdm_drain() {
#if QATTN_HAS_TDM
#if __has_builtin(__builtin_amdgcn_s_wait_tensorcnt)
  __builtin_amdgcn_s_wait_tensorcnt(0);
#endif
#endif
}

__global__ __launch_bounds__(256) void qattn_absmax_kernel(
    const float* __restrict__ x, int n, float* __restrict__ out)
{
  __shared__ float red[256];
  float m = 0.0f;
  const int n4 = n >> 2;
  const float4* x4 = (const float4*)x;
  for (int i = blockIdx.x * 256 + threadIdx.x; i < n4; i += gridDim.x * 256) {
    const float4 v = x4[i];
    m = fmaxf(m, fmaxf(fmaxf(fabsf(v.x), fabsf(v.y)),
                       fmaxf(fabsf(v.z), fabsf(v.w))));
  }
  red[threadIdx.x] = m;
  __syncthreads();
  for (int s = 128; s > 0; s >>= 1) {
    if (threadIdx.x < s)
      red[threadIdx.x] = fmaxf(red[threadIdx.x], red[threadIdx.x + s]);
    __syncthreads();
  }
  if (threadIdx.x == 0)
    atomicMax((unsigned int*)out, __float_as_uint(red[0]));
}

__global__ __launch_bounds__(256) void qattn_quant_kernel(
    const float* __restrict__ x, const float* __restrict__ amax,
    signed char* __restrict__ q, int n)
{
  const float scale = fmaxf(amax[0] * (1.0f / 127.0f), 1e-8f);
  const float inv = 1.0f / scale;
  const int n4 = n >> 2;
  const float4* x4 = (const float4*)x;
  int* q4 = (int*)q;
  for (int i = blockIdx.x * 256 + threadIdx.x; i < n4; i += gridDim.x * 256) {
    const float4 v = x4[i];
    const int a = (int)fminf(fmaxf(rintf(v.x * inv), -127.0f), 127.0f);
    const int b = (int)fminf(fmaxf(rintf(v.y * inv), -127.0f), 127.0f);
    const int c = (int)fminf(fmaxf(rintf(v.z * inv), -127.0f), 127.0f);
    const int d = (int)fminf(fmaxf(rintf(v.w * inv), -127.0f), 127.0f);
    q4[i] = (a & 0xFF) | ((b & 0xFF) << 8) | ((c & 0xFF) << 16) | (d << 24);
  }
}

__global__ __launch_bounds__(128) void qattn_gemm_iu8_kernel(
    const signed char* __restrict__ X, const signed char* __restrict__ Wm,
    const float* __restrict__ amax_x, const float* __restrict__ amax_w,
    const float* __restrict__ bias, void* __restrict__ out,
    int M, int N, int K, int mode)
{
  extern __shared__ char lds[];

  const int tid  = threadIdx.x;
  const int lane = tid & 31;
  const int wave = tid >> 5;
  const int lo = lane & 15, hi = lane >> 4;

  const int mtiles = M >> 5;
  const int mt = blockIdx.x % mtiles;
  const int ng = blockIdx.x / mtiles;
  const int nt = ng * 4 + wave;

  const signed char* aslab = X + (size_t)(mt * 32) * K;
  const signed char* wrow0 = Wm + (size_t)(nt * 32 + lo) * K;
  const signed char* wrow1 = wrow0 + (size_t)16 * K;

  if (wave == 0)
    qattn_tdm_warm(Wm + (size_t)(ng * 128) * K, K, 128u);

  {
    const signed char* src = aslab + (size_t)(tid >> 2) * K + ((tid & 3) << 4);
    *(int4*)(lds + (tid << 4)) = *(const int4*)src;
  }
  __syncthreads();

  v8i acc00 = {}, acc01 = {}, acc10 = {}, acc11 = {};
  for (int k0 = 0; k0 < K; k0 += 64) {
    const int cur = (k0 >> 6) & 1;
    const char* abuf = lds + cur * 2048;

    if (k0 + 64 < K) {
      const signed char* src =
          aslab + (size_t)(tid >> 2) * K + (k0 + 64) + ((tid & 3) << 4);
      *(int4*)(lds + ((cur ^ 1) * 2048) + (tid << 4)) = *(const int4*)src;
    }
    __builtin_prefetch(wrow0 + k0 + 64, 0, 1);
    __builtin_prefetch(wrow1 + k0 + 64, 0, 1);

    U8I a0, a1, b0, b1;
#pragma unroll
    for (int j = 0; j < 8; ++j) {
      const int ka = ((j >> 1) << 4) + ((j & 1) << 2) + (hi << 3);
      a0.i[j] = *(const int*)(abuf + lo * 64 + ka);
      a1.i[j] = *(const int*)(abuf + (16 + lo) * 64 + ka);
      const int kb = k0 + ((j >> 1) << 4) + ((j & 1) << 2) + (hi << 3);
      b0.i[j] = *(const int*)(wrow0 + kb);
      b1.i[j] = *(const int*)(wrow1 + kb);
    }
    acc00 = __builtin_amdgcn_wmma_i32_16x16x64_iu8(true, a0.v, true, b0.v, acc00, false, false);
    acc01 = __builtin_amdgcn_wmma_i32_16x16x64_iu8(true, a0.v, true, b1.v, acc01, false, false);
    acc10 = __builtin_amdgcn_wmma_i32_16x16x64_iu8(true, a1.v, true, b0.v, acc10, false, false);
    acc11 = __builtin_amdgcn_wmma_i32_16x16x64_iu8(true, a1.v, true, b1.v, acc11, false, false);

    __syncthreads();
  }
  qattn_tdm_drain();

  const float s = fmaxf(amax_x[0] * (1.0f / 127.0f), 1e-8f) *
                  fmaxf(amax_w[0] * (1.0f / 127.0f), 1e-8f);
  const v8i accs[2][2] = { { acc00, acc01 }, { acc10, acc11 } };

#pragma unroll
  for (int ti = 0; ti < 2; ++ti) {
#pragma unroll
    for (int tj = 0; tj < 2; ++tj) {
      const int n = nt * 32 + tj * 16 + lo;
      const float bn = bias[n];
      const v8i acc = accs[ti][tj];
      if (mode == 0) {
        float* o = (float*)out;
#pragma unroll
        for (int r = 0; r < 8; ++r) {
          const int m = mt * 32 + ti * 16 + (hi << 3) + r;
          o[(size_t)m * N + n] = (float)acc[r] * s + bn;
        }
      } else {
        _Float16* o = (_Float16*)out;
        const int h = n >> 6;
        const int d = n & 63;
#pragma unroll
        for (int r = 0; r < 8; ++r) {
          const int m  = mt * 32 + ti * 16 + (hi << 3) + r;
          const int bb = m >> 11;
          const int ss = m & (S_ - 1);
          const float y = (float)acc[r] * s + bn;
          if (mode == 1)
            o[((((size_t)bb * H_ + h) * S_ + ss) << 6) + d] = (_Float16)y;
          else
            o[(((size_t)bb * H_ + h) * HD_ + d) * S_ + ss] = (_Float16)y;
        }
      }
    }
  }
}

__global__ __launch_bounds__(128) void qattn_attn_kernel(
    const _Float16* __restrict__ Q, const _Float16* __restrict__ Kt,
    const _Float16* __restrict__ Vt, const float* __restrict__ mask,
    float* __restrict__ attn_out)
{
  extern __shared__ float smem[];
  float* sc     = smem;
  float* red    = smem + 16 * S_;
  float* rowmax = red + 16 * 8;
  float* rowinv = rowmax + 16;

  const int tid  = threadIdx.x;
  const int lane = tid & 31, wave = tid >> 5;
  const int lo = lane & 15, hi = lane >> 4;

  const int qt = blockIdx.x & (S_ / 16 - 1);
  const int bh = blockIdx.x >> 7;
  const int h = bh & (H_ - 1);
  const int b = bh >> 4;

  const _Float16* Qh = Q  + (size_t)bh * S_ * HD_;
  const _Float16* Kh = Kt + (size_t)bh * S_ * HD_;
  const _Float16* Vh = Vt + (size_t)bh * S_ * HD_;

  U16H aq0, aq1;
  {
    const int* qrow = (const int*)(Qh + (size_t)(qt * 16 + lo) * HD_);
#pragma unroll
    for (int j = 0; j < 8; ++j) {
      const int k = ((j >> 2) << 4) + ((j & 3) << 1) + (hi << 3);
      aq0.i[j] = qrow[k >> 1];
      aq1.i[j] = qrow[(k + 32) >> 1];
    }
  }

  for (int kt = wave; kt < S_ / 16; kt += 4) {
    U16H b0, b1;
    const int* krow = (const int*)(Kh + (size_t)(kt * 16 + lo) * HD_);
#pragma unroll
    for (int j = 0; j < 8; ++j) {
      const int wi = ((j >> 2) << 3) + (j & 3) + (hi << 2);
      b0.i[j] = krow[wi];
      b1.i[j] = krow[wi + 16];
    }
    v8f cacc = {};
    cacc = __builtin_amdgcn_wmma_f32_16x16x32_f16(false, aq0.v, false, b0.v,
                                                  (short)0, cacc, false, false);
    cacc = __builtin_amdgcn_wmma_f32_16x16x32_f16(false, aq1.v, false, b1.v,
                                                  (short)0, cacc, false, false);
    const int key = kt * 16 + lo;
    const float mb = (1.0f - mask[b * S_ + key]) * -10000.0f;
#pragma unroll
    for (int r = 0; r < 8; ++r)
      sc[(size_t)((hi << 3) + r) * S_ + key] = cacc[r] * 0.125f + mb;
  }
  __syncthreads();

  {
    const int r = tid & 15, ch = tid >> 4;
    const int c0 = ch * (S_ / 8), c1 = c0 + S_ / 8;
    float m = -3.0e38f;
    for (int c = c0; c < c1; ++c) m = fmaxf(m, sc[(size_t)r * S_ + c]);
    red[r * 8 + ch] = m;
    __syncthreads();
    if (ch == 0) {
      float mm = red[r * 8];
#pragma unroll
      for (int t = 1; t < 8; ++t) mm = fmaxf(mm, red[r * 8 + t]);
      rowmax[r] = mm;
    }
    __syncthreads();
    const float rm = rowmax[r];
    float sum = 0.0f;
    for (int c = c0; c < c1; ++c) {
      const float e = expf(sc[(size_t)r * S_ + c] - rm);
      sc[(size_t)r * S_ + c] = e;
      sum += e;
    }
    red[r * 8 + ch] = sum;
    __syncthreads();
    if (ch == 0) {
      float tt = red[r * 8];
#pragma unroll
      for (int t = 1; t < 8; ++t) tt += red[r * 8 + t];
      rowinv[r] = 1.0f / tt;
    }
    __syncthreads();
  }

  const int d0 = wave * 16;
  const int* vrow = (const int*)(Vh + (size_t)(d0 + lo) * S_);
  v8f acc = {};
  for (int k0 = 0; k0 < S_; k0 += 32) {
    U16H ap, bv;
#pragma unroll
    for (int j = 0; j < 8; ++j) {
      const int ka = k0 + ((j >> 2) << 4) + ((j & 3) << 1) + (hi << 3);
      ap.h[2 * j]     = (_Float16)sc[(size_t)lo * S_ + ka];
      ap.h[2 * j + 1] = (_Float16)sc[(size_t)lo * S_ + ka + 1];
      const int kb = k0 + ((j >> 2) << 4) + ((j & 3) << 1) + (hi << 3);
      bv.i[j] = vrow[kb >> 1];
    }
    acc = __builtin_amdgcn_wmma_f32_16x16x32_f16(false, ap.v, false, bv.v,
                                                 (short)0, acc, false, false);
  }
#pragma unroll
  for (int r = 0; r < 8; ++r) {
    const int m = (hi << 3) + r;
    const int srow = qt * 16 + m;
    attn_out[((size_t)b * S_ + srow) * D_ + h * HD_ + d0 + lo] =
        acc[r] * rowinv[m];
  }
}

extern "C" void kernel_launch(void* const* d_in, const int* in_sizes, int n_in,
                              void* d_out, int out_size, void* d_ws, size_t ws_size,
                              hipStream_t stream) {
  (void)in_sizes; (void)n_in; (void)out_size; (void)ws_size;
  const float* hidden = (const float*)d_in[0];
  const float* mask   = (const float*)d_in[1];
  const float* Wq = (const float*)d_in[2];  const float* bq = (const float*)d_in[3];
  const float* Wk = (const float*)d_in[4];  const float* bk = (const float*)d_in[5];
  const float* Wv = (const float*)d_in[6];  const float* bv = (const float*)d_in[7];
  const float* Wo = (const float*)d_in[8];  const float* bo = (const float*)d_in[9];
  float* out = (float*)d_out;

  char* w = (char*)d_ws;
  float* amax = (float*)w;              w += 256;
  signed char* xq  = (signed char*)w;   w += (size_t)M_ * D_;
  signed char* wqq = (signed char*)w;   w += (size_t)D_ * D_;
  signed char* wkq = (signed char*)w;   w += (size_t)D_ * D_;
  signed char* wvq = (signed char*)w;   w += (size_t)D_ * D_;
  signed char* woq = (signed char*)w;   w += (size_t)D_ * D_;
  signed char* aq  = (signed char*)w;   w += (size_t)M_ * D_;
  _Float16* Qh = (_Float16*)w;          w += (size_t)M_ * D_ * 2;
  _Float16* Kh = (_Float16*)w;          w += (size_t)M_ * D_ * 2;
  _Float16* Vh = (_Float16*)w;          w += (size_t)M_ * D_ * 2;
  float* attn = (float*)w;              w += (size_t)M_ * D_ * 4;

  hipMemsetAsync(amax, 0, 256, stream);

  qattn_absmax_kernel<<<256, 256, 0, stream>>>(hidden, M_ * D_, amax + 0);
  qattn_absmax_kernel<<<128, 256, 0, stream>>>(Wq, D_ * D_, amax + 1);
  qattn_absmax_kernel<<<128, 256, 0, stream>>>(Wk, D_ * D_, amax + 2);
  qattn_absmax_kernel<<<128, 256, 0, stream>>>(Wv, D_ * D_, amax + 3);
  qattn_absmax_kernel<<<128, 256, 0, stream>>>(Wo, D_ * D_, amax + 4);

  qattn_quant_kernel<<<512, 256, 0, stream>>>(hidden, amax + 0, xq, M_ * D_);
  qattn_quant_kernel<<<256, 256, 0, stream>>>(Wq, amax + 1, wqq, D_ * D_);
  qattn_quant_kernel<<<256, 256, 0, stream>>>(Wk, amax + 2, wkq, D_ * D_);
  qattn_quant_kernel<<<256, 256, 0, stream>>>(Wv, amax + 3, wvq, D_ * D_);
  qattn_quant_kernel<<<256, 256, 0, stream>>>(Wo, amax + 4, woq, D_ * D_);

  const int gemm_blocks = (M_ / 32) * (D_ / 128);
  qattn_gemm_iu8_kernel<<<gemm_blocks, 128, GEMM_LDS_BYTES, stream>>>(
      xq, wqq, amax + 0, amax + 1, bq, (void*)Qh, M_, D_, D_, 1);
  qattn_gemm_iu8_kernel<<<gemm_blocks, 128, GEMM_LDS_BYTES, stream>>>(
      xq, wkq, amax + 0, amax + 2, bk, (void*)Kh, M_, D_, D_, 1);
  qattn_gemm_iu8_kernel<<<gemm_blocks, 128, GEMM_LDS_BYTES, stream>>>(
      xq, wvq, amax + 0, amax + 3, bv, (void*)Vh, M_, D_, D_, 2);

  const size_t smem_bytes = (size_t)(16 * S_ + 16 * 8 + 16 + 16) * sizeof(float);
  hipFuncSetAttribute((const void*)qattn_attn_kernel,
                      hipFuncAttributeMaxDynamicSharedMemorySize,
                      (int)smem_bytes);
  qattn_attn_kernel<<<B_ * H_ * (S_ / 16), 128, smem_bytes, stream>>>(
      Qh, Kh, Vh, mask, attn);

  qattn_absmax_kernel<<<256, 256, 0, stream>>>(attn, M_ * D_, amax + 5);
  qattn_quant_kernel<<<512, 256, 0, stream>>>(attn, amax + 5, aq, M_ * D_);
  qattn_gemm_iu8_kernel<<<gemm_blocks, 128, GEMM_LDS_BYTES, stream>>>(
      aq, woq, amax + 5, amax + 4, bo, (void*)out, M_, D_, D_, 0);
}
